// GineEncoder_12180527252067
// MI455X (gfx1250) — matrix operand order repaired and checked by reading, not compiled, not run on hardware
//
#include <hip/hip_runtime.h>
#include <hip/hip_bf16.h>

// GINE encoder for MI455X (gfx1250), wave32 + WMMA bf16 + TDM.
// N=50000 nodes, E=256000 edges, G=1024 graphs, D=300 (padded to P=320), L=5.
// All D x D GEMMs: v_wmma_f32_16x16x32_bf16, fp32 accumulate.
// Packed bf16 weight matrix (200KB) is DMA'd into LDS once per block via the
// Tensor Data Mover (tensor_load_to_lds + s_wait_tensorcnt) and reused across
// 5 row-tiles (80 rows), cutting B L2 traffic 5x. LDS: 200KB B + 10KB A.
// Message passing / LayerNorm / pooling stay fp32 (HBM-bound path).
// Workspace requirement: ~329 MB.

#define N_NODES 50000
#define N_EDGES 256000
#define N_GRAPH 1024
#define DIM     300
#define PD      320      // padded feature stride: 10 K-blocks of 32, 20 col-tiles of 16
#define LN_EPS  1e-5f

#define MATB      (20 * 10 * 32 * 16)       // bf16 elems per packed matrix = 102400
#define MAT_BYTES (MATB * 2)                // 204800 B
#define SMEM_SZ   (MAT_BYTES + 16 * PD * 2) // 204800 + 10240 = 215040 B

typedef __attribute__((ext_vector_type(16))) __bf16 bf16x16;
typedef __attribute__((ext_vector_type(8)))  float  f32x8;
typedef __attribute__((ext_vector_type(4)))  unsigned int u32x4;
typedef __attribute__((ext_vector_type(8)))  int i32x8;
typedef __attribute__((ext_vector_type(4)))  int i32x4;

#if defined(__has_builtin)
#if __has_builtin(__builtin_amdgcn_tensor_load_to_lds)
#define HAS_TDM 1
#endif
#if __has_builtin(__builtin_amdgcn_s_wait_tensorcnt)
#define HAS_TCNT 1
#endif
#endif

__device__ __forceinline__ void wait_tensorcnt0() {
#if defined(HAS_TCNT)
  __builtin_amdgcn_s_wait_tensorcnt(0);
#else
  asm volatile("s_wait_tensorcnt 0x0" ::: "memory");
#endif
}

// Issue one TDM transfer: MAT_BYTES contiguous bytes global -> LDS.
// 1-D tile of 25600 x 8-byte elements (fits 16-bit tile_dim0 field).
__device__ __forceinline__ void tdm_load_matrix(const __bf16* gsrc, void* lds_dst) {
#if defined(HAS_TDM)
  unsigned long long ga = (unsigned long long)(uintptr_t)gsrc;
  unsigned lds_addr = (unsigned)(uintptr_t)lds_dst;   // low 32 bits = LDS byte offset
  const unsigned QW = MAT_BYTES / 8;                  // 25600
  u32x4 g0 = { 1u,                                    // count=1, user descriptor
               lds_addr,
               (unsigned)(ga & 0xFFFFFFFFu),
               (unsigned)((ga >> 32) & 0x01FFFFFFu) | (2u << 30) };  // type=2
  i32x8 g1 = { (int)(3u << 16),                       // data_size=3 (8B), mask/flags=0
               (int)((QW & 0xFFFFu) << 16),           // tensor_dim0 lo16
               (int)(((QW >> 16) & 0xFFFFu) | (1u << 16)), // tensor_dim0 hi16 | tensor_dim1 lo16=1
               (int)((QW & 0xFFFFu) << 16),           // tensor_dim1 hi=0 | tile_dim0
               0,                                     // tile_dim1=0, tile_dim2=0
               (int)QW,                               // tensor_dim0_stride lo32
               0, 0 };
  i32x4 g2 = { 0, 0, 0, 0 };
  i32x4 g3 = { 0, 0, 0, 0 };
#if __clang_major__ >= 23
  i32x8 g4 = { 0, 0, 0, 0, 0, 0, 0, 0 };
  __builtin_amdgcn_tensor_load_to_lds(g0, g1, g2, g3, g4, 0);
#else
  __builtin_amdgcn_tensor_load_to_lds(g0, g1, g2, g3, 0);
#endif
#else
  (void)gsrc; (void)lds_dst;
#endif
}

struct GP {
  const float*  f0;   // mode0: chirality   mode1: edge_attr  mode2: h (fp32 A)
  const float*  f1;   // mode0/1: first-layer weight [2|3, DIM]
  const float*  f2;   // mode0/1: first-layer bias [DIM]
  const float*  f3;   // mode0: formal_charge
  const __bf16* abf;  // mode3/4: bf16 A source
  const __bf16* bpk;  // packed B panels (WMMA lane layout)
  const float*  bias; // epilogue bias [DIM]
  const float*  addx; // mode3: residual x
  const int*    z;    // mode0: atom ids
  const float*  emb;  // mode0: atom embedding [VOCAB, DIM]
  float*        outf;
  __bf16*       outb;
};

// ---- pack one [DIM,DIM] fp32 weight into WMMA B-fragment panels ------------
// layout: [jt(20)][kb(10)][lane(32)][e(16)] bf16, lane<16 -> K kb*32+0..15,
// lane>=16 -> K kb*32+16..31 ; N = jt*16 + (lane&15). Pad with zeros.
__global__ __launch_bounds__(256) void k_pack_b(const float* __restrict__ w,
                                                __bf16* __restrict__ dst) {
  int i = blockIdx.x * 256 + threadIdx.x;
  if (i >= MATB) return;
  int e    = i & 15;
  int lane = (i >> 4) & 31;
  int kb   = (i >> 9) % 10;
  int jt   = i / (16 * 32 * 10);
  int k = kb * 32 + ((lane >> 4) ? 16 : 0) + e;
  int n = jt * 16 + (lane & 15);
  float v = (k < DIM && n < DIM) ? w[k * DIM + n] : 0.0f;
  dst[i] = (__bf16)v;
}

// ---- fused GEMM: NT row-tiles x full width, B resident in LDS via TDM ------
// MODE 0: A = relu([chir,fc] @ nap_w1 + b1);      out x  = acc + b2 + emb[z]   (f32)
// MODE 1: A = relu(edge_attr @ ee_w1 + b1);       out e  = acc + b2            (bf16)
// MODE 2: A = bf16(h fp32);                       out t  = relu(acc + b1_l)    (bf16)
// MODE 3: A = t (bf16);                           out y  = x + relu(acc + b2_l)(f32)
// MODE 4: A = pooled (bf16);                      out    = acc + pool_b  [G,DIM] f32
template <int MODE, int NT>
__global__ __launch_bounds__(256) void k_gemm(GP p) {
  extern __shared__ char smem[];
  __bf16* Bs = (__bf16*)smem;                       // 204800 B packed weights
  __bf16* As = (__bf16*)(smem + MAT_BYTES);         // 10240 B row-panel

  const int tid  = threadIdx.x;
  const int wv   = tid >> 5, lane = tid & 31;
  const int sel  = lane >> 4, mrow = lane & 15;

#if defined(HAS_TDM)
  if (tid < 32) tdm_load_matrix(p.bpk, Bs);         // wave 0 issues the DMA
#else
  for (int i = tid; i < MAT_BYTES / 16; i += 256)
    ((uint4*)Bs)[i] = ((const uint4*)p.bpk)[i];
#endif

  for (int rt = 0; rt < NT; ++rt) {
    const int tile = blockIdx.x * NT + rt;
    __syncthreads();                                // WAR: prior reads of As done
    // ---- stage A (16 x 320 bf16) ----
    for (int i = tid; i < 16 * PD; i += 256) {
      int r = i / PD, c = i - r * PD;
      int row = tile * 16 + r;
      if (MODE == 3 || MODE == 4) {
        As[i] = p.abf[(size_t)row * PD + c];
      } else {
        float v = 0.f;
        if (MODE == 0) {
          if (c < DIM) {
            v = p.f0[row] * p.f1[c] + p.f3[row] * p.f1[DIM + c] + p.f2[c];
            v = fmaxf(v, 0.f);
          }
        } else if (MODE == 1) {
          if (c < DIM) {
            v = p.f0[row * 3 + 0] * p.f1[c] + p.f0[row * 3 + 1] * p.f1[DIM + c] +
                p.f0[row * 3 + 2] * p.f1[2 * DIM + c] + p.f2[c];
            v = fmaxf(v, 0.f);
          }
        } else {                                    // MODE 2: h already zero-padded
          v = p.f0[(size_t)row * PD + c];
        }
        As[i] = (__bf16)v;
      }
    }
#if defined(HAS_TDM)
    if (rt == 0 && tid < 32) wait_tensorcnt0();     // B DMA complete before use
#endif
    __syncthreads();

    // ---- WMMA: each wave owns col-tiles jt = wv, wv+8, wv+16 ----
    for (int jt = wv; jt < 20; jt += 8) {
      f32x8 acc = {0.f, 0.f, 0.f, 0.f, 0.f, 0.f, 0.f, 0.f};
      const __bf16* bbase = Bs + (size_t)jt * (10 * 32 * 16);
      for (int kb = 0; kb < 10; ++kb) {
        // A frag: lane<16 row=lane K=kb*32+0..15; lane>=16 -> K=kb*32+16..31
        bf16x16 a;
        const __bf16* ap = &As[mrow * PD + kb * 32 + sel * 16];
        reinterpret_cast<uint4*>(&a)[0] = *reinterpret_cast<const uint4*>(ap);
        reinterpret_cast<uint4*>(&a)[1] = *reinterpret_cast<const uint4*>(ap + 8);
        // B frag: one contiguous 32B chunk per lane (pre-packed, LDS-resident)
        bf16x16 b;
        const __bf16* bp = bbase + (size_t)(kb * 32 + lane) * 16;
        reinterpret_cast<uint4*>(&b)[0] = *reinterpret_cast<const uint4*>(bp);
        reinterpret_cast<uint4*>(&b)[1] = *reinterpret_cast<const uint4*>(bp + 8);
        acc = __builtin_amdgcn_wmma_f32_16x16x32_bf16(false, a, false, b,
                                                      (short)0, acc, false, false);
      }
      // C/D layout: lane<16 -> N=lane, M=v ; lane>=16 -> N=lane-16, M=8+v
      int col = jt * 16 + mrow;
      int rb  = tile * 16 + sel * 8;
      for (int v = 0; v < 8; ++v) {
        int row = rb + v;
        float val = acc[v];
        if (MODE == 0) {
          if (col < DIM) val += p.bias[col] + p.emb[(size_t)p.z[row] * DIM + col];
          p.outf[(size_t)row * PD + col] = val;
        } else if (MODE == 1) {
          if (col < DIM) val += p.bias[col];
          p.outb[(size_t)row * PD + col] = (__bf16)val;
        } else if (MODE == 2) {
          if (col < DIM) val += p.bias[col];
          p.outb[(size_t)row * PD + col] = (__bf16)fmaxf(val, 0.f);
        } else if (MODE == 3) {
          if (col < DIM) val += p.bias[col];
          p.outf[(size_t)row * PD + col] =
              p.addx[(size_t)row * PD + col] + fmaxf(val, 0.f);
        } else {                                    // MODE 4
          if (col < DIM) p.outf[(size_t)row * DIM + col] = val + p.bias[col];
        }
      }
    }
  }
}

// ---- h = x (aggregation initialised with self term) ------------------------
__global__ __launch_bounds__(256) void k_copy(const float* __restrict__ s,
                                              float* __restrict__ d, int n) {
  int i = blockIdx.x * 256 + threadIdx.x;
  if (i < n) d[i] = s[i];
}

// ---- message pass: h[dst] += relu(x[src] + e) ; 1 wave per edge ------------
__global__ __launch_bounds__(256) void k_msg(const int* __restrict__ ei,
                                             const __bf16* __restrict__ e,
                                             const float* __restrict__ x,
                                             float* __restrict__ h) {
  int eid  = blockIdx.x * 8 + (threadIdx.x >> 5);
  int lane = threadIdx.x & 31;
  int s = ei[eid], d = ei[N_EDGES + eid];
  const float*  xs = x + (size_t)s * PD;
  const __bf16* es = e + (size_t)eid * PD;
  float*        hd = h + (size_t)d * PD;
  for (int c = lane; c < DIM; c += 32) {
    float m = xs[c] + (float)es[c];
    if (m > 0.f) atomicAdd(&hd[c], m);
  }
}

// ---- residual LayerNorm: x = LN(y) * g + b ; 1 wave per row ----------------
__global__ __launch_bounds__(256) void k_ln(const float* __restrict__ y,
                                            float* __restrict__ x,
                                            const float* __restrict__ g,
                                            const float* __restrict__ b) {
  int row  = blockIdx.x * 8 + (threadIdx.x >> 5);
  int lane = threadIdx.x & 31;
  const float* yr = y + (size_t)row * PD;
  float v[10];
  float s = 0.f, ss = 0.f;
  int k = 0;
  for (int c = lane; c < DIM; c += 32) { float t = yr[c]; v[k++] = t; s += t; ss += t * t; }
  for (int o = 16; o; o >>= 1) { s += __shfl_xor(s, o, 32); ss += __shfl_xor(ss, o, 32); }
  float mu  = s * (1.f / DIM);
  float var = ss * (1.f / DIM) - mu * mu;
  float rs  = rsqrtf(var + LN_EPS);
  float* xr = x + (size_t)row * PD;
  k = 0;
  for (int c = lane; c < DIM; c += 32) xr[c] = (v[k++] - mu) * rs * g[c] + b[c];
  for (int c = DIM + lane; c < PD; c += 32) xr[c] = 0.f;   // keep pad columns zero
}

// ---- pooling ---------------------------------------------------------------
__global__ __launch_bounds__(256) void k_zero(float* s, float* c) {
  int i = blockIdx.x * 256 + threadIdx.x;
  if (i < N_GRAPH * PD) s[i] = 0.f;
  if (i < N_GRAPH) c[i] = 0.f;
}
__global__ __launch_bounds__(256) void k_pool_acc(const int* __restrict__ batch,
                                                  const float* __restrict__ x,
                                                  float* __restrict__ sums,
                                                  float* __restrict__ cnt) {
  int n    = blockIdx.x * 8 + (threadIdx.x >> 5);
  int lane = threadIdx.x & 31;
  int b = batch[n];
  const float* xr = x + (size_t)n * PD;
  float* sr = sums + (size_t)b * PD;
  for (int c = lane; c < DIM; c += 32) atomicAdd(&sr[c], xr[c]);
  if (lane == 0) atomicAdd(&cnt[b], 1.f);
}
__global__ __launch_bounds__(256) void k_pool_fin(const float* __restrict__ sums,
                                                  const float* __restrict__ cnt,
                                                  __bf16* __restrict__ pooled) {
  int i = blockIdx.x * 256 + threadIdx.x;
  if (i >= N_GRAPH * PD) return;
  int g = i / PD, c = i - g * PD;
  float cv = cnt[g];
  float v = (cv > 0.f && c < DIM) ? sums[i] / fmaxf(cv, 1.f) : 0.f;
  pooled[i] = (__bf16)v;
}

extern "C" void kernel_launch(void* const* d_in, const int* in_sizes, int n_in,
                              void* d_out, int out_size, void* d_ws, size_t ws_size,
                              hipStream_t stream) {
  (void)in_sizes; (void)n_in; (void)out_size; (void)ws_size;
  const int*   z     = (const int*)d_in[0];
  const float* chir  = (const float*)d_in[1];
  const float* fch   = (const float*)d_in[2];
  const int*   eidx  = (const int*)d_in[3];
  const float* eattr = (const float*)d_in[4];
  const int*   batch = (const int*)d_in[5];
  const float* emb   = (const float*)d_in[6];
  const float* nw1   = (const float*)d_in[7];
  const float* nb1   = (const float*)d_in[8];
  const float* nw2   = (const float*)d_in[9];
  const float* nb2   = (const float*)d_in[10];
  const float* ew1   = (const float*)d_in[11];
  const float* eb1   = (const float*)d_in[12];
  const float* ew2   = (const float*)d_in[13];
  const float* eb2   = (const float*)d_in[14];
  const float* cw1   = (const float*)d_in[15];
  const float* cb1   = (const float*)d_in[16];
  const float* cw2   = (const float*)d_in[17];
  const float* cb2   = (const float*)d_in[18];
  const float* lng   = (const float*)d_in[19];
  const float* lnb   = (const float*)d_in[20];
  const float* pw    = (const float*)d_in[21];
  const float* pb    = (const float*)d_in[22];
  float* out = (float*)d_out;

  // workspace carve-up (~329 MB)
  char* w = (char*)d_ws;
  __bf16* bpk    = (__bf16*)w; w += (size_t)13 * MAT_BYTES;
  float*  x      = (float*)w;  w += (size_t)N_NODES * PD * 4;
  float*  h      = (float*)w;  w += (size_t)N_NODES * PD * 4;     // also holds y
  __bf16* t      = (__bf16*)w; w += (size_t)N_NODES * PD * 2;
  __bf16* ebf    = (__bf16*)w; w += (size_t)N_EDGES * PD * 2;
  float*  sums   = (float*)w;  w += (size_t)N_GRAPH * PD * 4;
  float*  cnt    = (float*)w;  w += (size_t)N_GRAPH * 4;
  __bf16* pooled = (__bf16*)w;

  // allow 215 KB dynamic LDS (320 KB/WGP on gfx1250)
  hipFuncSetAttribute(reinterpret_cast<const void*>(&k_gemm<0, 5>),
                      hipFuncAttributeMaxDynamicSharedMemorySize, SMEM_SZ);
  hipFuncSetAttribute(reinterpret_cast<const void*>(&k_gemm<1, 5>),
                      hipFuncAttributeMaxDynamicSharedMemorySize, SMEM_SZ);
  hipFuncSetAttribute(reinterpret_cast<const void*>(&k_gemm<2, 5>),
                      hipFuncAttributeMaxDynamicSharedMemorySize, SMEM_SZ);
  hipFuncSetAttribute(reinterpret_cast<const void*>(&k_gemm<3, 5>),
                      hipFuncAttributeMaxDynamicSharedMemorySize, SMEM_SZ);
  hipFuncSetAttribute(reinterpret_cast<const void*>(&k_gemm<4, 4>),
                      hipFuncAttributeMaxDynamicSharedMemorySize, SMEM_SZ);

  dim3 B(256);
  const int packG = (MATB + 255) / 256;

  // pack all 13 DxD weight matrices into WMMA-B layout (bf16)
  k_pack_b<<<packG, B, 0, stream>>>(nw2, bpk + 0 * MATB);
  k_pack_b<<<packG, B, 0, stream>>>(ew2, bpk + 1 * MATB);
  for (int l = 0; l < 5; ++l)
    k_pack_b<<<packG, B, 0, stream>>>(cw1 + (size_t)l * DIM * DIM, bpk + (size_t)(2 + l) * MATB);
  for (int l = 0; l < 5; ++l)
    k_pack_b<<<packG, B, 0, stream>>>(cw2 + (size_t)l * DIM * DIM, bpk + (size_t)(7 + l) * MATB);
  k_pack_b<<<packG, B, 0, stream>>>(pw, bpk + 12 * MATB);

  GP p;
  // node init: x = atom_emb[z] + MLP(chir, fc)
  p = GP{}; p.f0 = chir; p.f1 = nw1; p.f2 = nb1; p.f3 = fch;
  p.bpk = bpk + 0 * MATB; p.bias = nb2; p.z = z; p.emb = emb; p.outf = x;
  k_gemm<0, 5><<<N_NODES / 80, B, SMEM_SZ, stream>>>(p);

  // edge encoder: e = linear(relu(linear(edge_attr)))  (stored bf16)
  p = GP{}; p.f0 = eattr; p.f1 = ew1; p.f2 = eb1;
  p.bpk = bpk + 1 * MATB; p.bias = eb2; p.outb = ebf;
  k_gemm<1, 5><<<N_EDGES / 80, B, SMEM_SZ, stream>>>(p);

  for (int l = 0; l < 5; ++l) {
    k_copy<<<(N_NODES * PD + 255) / 256, B, 0, stream>>>(x, h, N_NODES * PD);
    k_msg<<<N_EDGES / 8, B, 0, stream>>>(eidx, ebf, x, h);
    p = GP{}; p.f0 = h; p.bpk = bpk + (size_t)(2 + l) * MATB; p.bias = cb1 + l * DIM; p.outb = t;
    k_gemm<2, 5><<<N_NODES / 80, B, SMEM_SZ, stream>>>(p);
    p = GP{}; p.abf = t; p.bpk = bpk + (size_t)(7 + l) * MATB; p.bias = cb2 + l * DIM;
    p.addx = x; p.outf = h;                          // h now holds y
    k_gemm<3, 5><<<N_NODES / 80, B, SMEM_SZ, stream>>>(p);
    k_ln<<<N_NODES / 8, B, 0, stream>>>(h, x, lng + l * DIM, lnb + l * DIM);
  }

  k_zero<<<(N_GRAPH * PD + 255) / 256, B, 0, stream>>>(sums, cnt);
  k_pool_acc<<<N_NODES / 8, B, 0, stream>>>(batch, x, sums, cnt);
  k_pool_fin<<<(N_GRAPH * PD + 255) / 256, B, 0, stream>>>(sums, cnt, pooled);

  p = GP{}; p.abf = pooled; p.bpk = bpk + 12 * MATB; p.bias = pb; p.outf = out;
  k_gemm<4, 4><<<N_GRAPH / 64, B, SMEM_SZ, stream>>>(p);
}
